// AdaptiveNoiseScheduler_3590592659664
// MI455X (gfx1250) — matrix operand order repaired and checked by reading, not compiled, not run on hardware
//
#include <hip/hip_runtime.h>
#include <hip/hip_bf16.h>
#include <math.h>

// ---------------------------------------------------------------------------
// Problem dims (fixed by the reference): B=4, S=4096, E=1024
//   combined : (16384 x 2048)  bf16
//   GEMM1    : combined @ W1t  -> x1 (16384 x 1024) bf16, exact-gelu epilogue
//   GEMM2    : x1       @ W2t  -> x2 (16384 x 512)  bf16, exact-gelu epilogue
//   head     : sigmoid(x2 @ W3 + b3) -> timestep math -> int32 (16384)
// ---------------------------------------------------------------------------

typedef __attribute__((ext_vector_type(16))) __bf16 v16bf;
typedef __attribute__((ext_vector_type(8)))  float  v8f;

struct FragI { int4 x; int4 y; };   // 32 bytes == one v16bf

#define BB 4
#define SS 4096
#define EE 1024
#define NCHUNK 32
#define CHLEN 128          // SS / NCHUNK
#define MTOT (BB * SS)     // 16384 rows

// ---------------- Stage 1: per-chunk sums for the S-cumsum --------------------
__global__ __launch_bounds__(256) void chunk_sums(const float* __restrict__ h,
                                                  float* __restrict__ csum) {
  const int e = blockIdx.x * 256 + threadIdx.x;   // 0..1023 (grid.x = 4)
  const int c = blockIdx.y;                       // chunk
  const int b = blockIdx.z;                       // batch
  const float* p = h + ((size_t)b * SS + (size_t)c * CHLEN) * EE + e;
  float s = 0.f;
#pragma unroll 4
  for (int i = 0; i < CHLEN; ++i) s += p[(size_t)i * EE];
  csum[((size_t)b * NCHUNK + c) * EE + e] = s;
}

// ---------------- Stage 2: build combined = [h | 0.5*(fwd+bwd)] in bf16 -------
__global__ __launch_bounds__(256) void build_combined(const float* __restrict__ h,
                                                      const float* __restrict__ csum,
                                                      __bf16* __restrict__ comb) {
  const int e = blockIdx.x * 256 + threadIdx.x;
  const int c = blockIdx.y;
  const int b = blockIdx.z;
  float prefix = 0.f, total = 0.f;
#pragma unroll
  for (int j = 0; j < NCHUNK; ++j) {
    float v = csum[((size_t)b * NCHUNK + j) * EE + e];
    total += v;
    if (j < c) prefix += v;
  }
  float running = prefix;
  const float* hp = h    + ((size_t)b * SS + (size_t)c * CHLEN) * EE + e;
  __bf16*      cp = comb + ((size_t)b * SS + (size_t)c * CHLEN) * (2 * EE) + e;
  for (int i = 0; i < CHLEN; ++i) {
    const int s = c * CHLEN + i;
    const float hv = hp[(size_t)i * EE];
    const float past = running;
    running += hv;
    const float fwd = (s < SS - 1) ? (total - running) / (float)(SS - 1 - s) : 0.f;
    const float bwd = (s > 0) ? past / (float)s : 0.f;
    cp[(size_t)i * (2 * EE)]      = (__bf16)hv;
    cp[(size_t)i * (2 * EE) + EE] = (__bf16)(0.5f * (fwd + bwd));
  }
}

// ---------------- Weight transpose + fp32 -> bf16 (K-contiguous rows) ---------
// src: (R x C) fp32 row-major  ->  dst: (C x R) bf16 row-major
__global__ __launch_bounds__(256) void transpose_to_bf16(const float* __restrict__ src,
                                                         __bf16* __restrict__ dst,
                                                         int R, int C) {
  __shared__ float tile[32][33];
  const int tx = threadIdx.x, ty = threadIdx.y;       // block (32, 8)
  const int c0 = blockIdx.x * 32, r0 = blockIdx.y * 32;
#pragma unroll
  for (int i = 0; i < 4; ++i)
    tile[ty + i * 8][tx] = src[(size_t)(r0 + ty + i * 8) * C + (c0 + tx)];
  __syncthreads();
#pragma unroll
  for (int i = 0; i < 4; ++i)
    dst[(size_t)(c0 + ty + i * 8) * R + (r0 + tx)] = (__bf16)tile[tx][ty + i * 8];
}

// ---------------- bf16 WMMA GEMM + bias + exact GELU --------------------------
// C[M x N] = gelu(A[M x K] @ Bt[N x K]^T + bias)   (A, Bt bf16 row-major, K-major)
// Block tile 256x128, 8 waves = 4(M) x 2(N), wave tile 64x64 = 4x4 WMMA frags.
// Software pipeline: global->reg prefetch issued BEFORE the 16 WMMAs of the
// current K-step, reg->LDS stores sunk after, so L2/HBM latency overlaps math.
// Single non-unrolled loop keeps pressure < 256 VGPRs (no VGPR-MSB mode).
__global__ __launch_bounds__(256) void gemm_bf16_gelu(const __bf16* __restrict__ A,
                                                      const __bf16* __restrict__ Bt,
                                                      const float* __restrict__ bias,
                                                      __bf16* __restrict__ C,
                                                      int M, int N, int K) {
  // 80-byte (40-element) row pitch: spreads the 16 rows a half-wave touches
  // across distinct bank ranges for the b128 fragment loads.
  __shared__ __align__(16) __bf16 As[2][256 * 40];   // 2 x 20 KB
  __shared__ __align__(16) __bf16 Bs[2][128 * 40];   // 2 x 10 KB

  const int tid   = threadIdx.x;
  const int lane  = tid & 31;
  const int wave  = tid >> 5;
  const int waveM = wave >> 1;       // 0..3  -> 64-row strip of 256
  const int waveN = wave & 1;        // 0..1  -> 64-col strip of 128
  const int l16   = lane & 15;
  const bool hi16 = lane >= 16;
  const int blockM = blockIdx.y * 256;
  const int blockN = blockIdx.x * 128;

  // Cooperative tile loader: per K-step the block stages A 256x32 (16 KB) and
  // B 128x32 (8 KB).  256 threads -> each does 2 A-rows (r, r+128) + 1 B-row,
  // 32 B (two b128) each.
  const int ldRow  = tid >> 1;            // 0..127
  const int ldHalf = (tid & 1) * 16;      // element offset 0 or 16 (=32 B)
  const __bf16* gA0 = A  + (size_t)(blockM + ldRow) * K + ldHalf;
  const __bf16* gA1 = gA0 + (size_t)128 * K;
  const __bf16* gB  = Bt + (size_t)(blockN + ldRow) * K + ldHalf;

  int4 st0, st1, st2, st3, st4, st5;      // in-flight stage registers

  auto load_regs = [&](int kt) {
    const int4* pa0 = (const int4*)(gA0 + (size_t)kt * 32);
    const int4* pa1 = (const int4*)(gA1 + (size_t)kt * 32);
    const int4* pb  = (const int4*)(gB  + (size_t)kt * 32);
    st0 = pa0[0]; st1 = pa0[1];
    st2 = pa1[0]; st3 = pa1[1];
    st4 = pb[0];  st5 = pb[1];
  };
  auto store_regs = [&](int buf) {
    int4* sa0 = (int4*)(&As[buf][ldRow * 40 + ldHalf]);
    int4* sa1 = (int4*)(&As[buf][(ldRow + 128) * 40 + ldHalf]);
    int4* sb  = (int4*)(&Bs[buf][ldRow * 40 + ldHalf]);
    sa0[0] = st0; sa0[1] = st1;
    sa1[0] = st2; sa1[1] = st3;
    sb[0]  = st4; sb[1]  = st5;
  };

  v8f acc[4][4] = {};

  auto compute = [&](int buf) {
    v16bf af[4], bfm[4];
    // A fragment (ISA 16-bit A 16x32 layout):
    //   lane<16 : row=l16, bytes [0,16) (K0..7)  and [32,48) (K16..23)
    //   lane>=16: row=l16, bytes [16,32)(K8..15) and [48,64) (K24..31)
#pragma unroll
    for (int mi = 0; mi < 4; ++mi) {
      const __bf16* p = &As[buf][(waveM * 64 + mi * 16 + l16) * 40 + (hi16 ? 8 : 0)];
      FragI f; f.x = *(const int4*)p; f.y = *(const int4*)(p + 16);
      af[mi] = __builtin_bit_cast(v16bf, f);
    }
    // B fragment (32x16): lane<16 -> K0..7,K16..23 of col N=lane; lane>=16 -> K8..15,K24..31.
#pragma unroll
    for (int ni = 0; ni < 4; ++ni) {
      const __bf16* p = &Bs[buf][(waveN * 64 + ni * 16 + l16) * 40 + (hi16 ? 8 : 0)];
      FragI f; f.x = *(const int4*)p; f.y = *(const int4*)(p + 16);
      bfm[ni] = __builtin_bit_cast(v16bf, f);
    }
#pragma unroll
    for (int mi = 0; mi < 4; ++mi)
#pragma unroll
      for (int ni = 0; ni < 4; ++ni)
        acc[mi][ni] = __builtin_amdgcn_wmma_f32_16x16x32_bf16(
            false, af[mi], false, bfm[ni], (short)0, acc[mi][ni], false, false);
  };

  // Prologue: stage 0 into buffer 0.
  load_regs(0);
  store_regs(0);
  __syncthreads();

  const int KT = K >> 5;                  // K-steps of 32
  for (int kt = 0; kt < KT; ++kt) {
    const int buf = kt & 1;
    const bool more = (kt + 1 < KT);
    if (more) load_regs(kt + 1);          // global loads issued first ...
    compute(buf);                         // ... overlap with 16 WMMAs ...
    if (more) store_regs(buf ^ 1);        // ... wait loads, then LDS stores
    __syncthreads();
  }

  // Epilogue: C/D layout -> lane<16: (M=r, N=lane); lane>=16: (M=r+8, N=lane-16)
#pragma unroll
  for (int mi = 0; mi < 4; ++mi) {
#pragma unroll
    for (int ni = 0; ni < 4; ++ni) {
      const int col = blockN + waveN * 64 + ni * 16 + l16;
      const float bs = bias[col];
      v8f a = acc[mi][ni];
#pragma unroll
      for (int r = 0; r < 8; ++r) {
        const int row = blockM + waveM * 64 + mi * 16 + r + (hi16 ? 8 : 0);
        const float v = a[r] + bs;
        const float g = 0.5f * v * (1.0f + erff(v * 0.70710678118654752f));
        C[(size_t)row * N + col] = (__bf16)g;
      }
    }
  }
}

// ---------------- Final head: dot(512) + sigmoid + timestep math --------------
__global__ __launch_bounds__(256) void final_head(const __bf16* __restrict__ x2,
                                                  const float* __restrict__ W3,
                                                  const float* __restrict__ b3,
                                                  const int* __restrict__ mask,
                                                  const int* __restrict__ tstep,
                                                  int* __restrict__ out) {
  const int wave = threadIdx.x >> 5;
  const int lane = threadIdx.x & 31;
  const int row  = blockIdx.x * 8 + wave;     // one wave per row
  const __bf16* xp = x2 + (size_t)row * 512;
  float sum = 0.f;
#pragma unroll
  for (int i = 0; i < 16; ++i) {
    const int k = i * 32 + lane;
    sum += (float)xp[k] * W3[k];
  }
#pragma unroll
  for (int off = 16; off; off >>= 1) sum += __shfl_down(sum, off, 32);
  if (lane == 0) {
    const float nf = 1.f / (1.f + expf(-(sum + b3[0])));
    float ad = (float)(*tstep) * (0.5f + nf);
    ad *= (1.f + (float)mask[row] * 0.3f);
    ad = fminf(fmaxf(ad, 0.f), 999.f);
    out[row] = (int)ad;
  }
}

// ---------------------------------------------------------------------------
extern "C" void kernel_launch(void* const* d_in, const int* in_sizes, int n_in,
                              void* d_out, int out_size, void* d_ws, size_t ws_size,
                              hipStream_t stream) {
  const float* h     = (const float*)d_in[0];
  const int*   mask  = (const int*)  d_in[1];
  const int*   tstep = (const int*)  d_in[2];
  const float* W1    = (const float*)d_in[3];
  const float* b1    = (const float*)d_in[4];
  const float* W2    = (const float*)d_in[5];
  const float* b2    = (const float*)d_in[6];
  const float* W3    = (const float*)d_in[7];
  const float* b3    = (const float*)d_in[8];
  int* out = (int*)d_out;

  char* ws = (char*)d_ws;
  // workspace layout (bytes)
  __bf16* comb = (__bf16*)(ws);                                 // 64 MB
  __bf16* W1t  = (__bf16*)(ws + 67108864ull);                   //  4 MB
  __bf16* W2t  = (__bf16*)(ws + 67108864ull + 4194304ull);      //  1 MB
  __bf16* x1   = (__bf16*)(ws + 72351744ull);                   // 32 MB
  __bf16* x2   = (__bf16*)(ws + 105906176ull);                  // 16 MB
  float*  csum = (float*) (ws + 122683392ull);                  // 512 KB

  // 1) chunked cumsum partials + combined build (streams h twice, HBM-bound)
  chunk_sums    <<<dim3(EE / 256, NCHUNK, BB), 256, 0, stream>>>(h, csum);
  build_combined<<<dim3(EE / 256, NCHUNK, BB), 256, 0, stream>>>(h, csum, comb);

  // 2) weights -> bf16, transposed so B-tiles are K-contiguous
  transpose_to_bf16<<<dim3(1024 / 32, 2048 / 32), dim3(32, 8), 0, stream>>>(W1, W1t, 2048, 1024);
  transpose_to_bf16<<<dim3(512 / 32, 1024 / 32), dim3(32, 8), 0, stream>>>(W2, W2t, 1024, 512);

  // 3) the two compute-bound GEMMs on the WMMA path (block tile 256x128)
  gemm_bf16_gelu<<<dim3(1024 / 128, MTOT / 256), 256, 0, stream>>>(comb, W1t, b1, x1, MTOT, 1024, 2048);
  gemm_bf16_gelu<<<dim3(512 / 128,  MTOT / 256), 256, 0, stream>>>(x1,   W2t, b2, x2, MTOT, 512,  1024);

  // 4) head + scheduler math -> int32
  final_head<<<MTOT / 8, 256, 0, stream>>>(x2, W3, b3, mask, tstep, out);
}
